// Gemma4SparseMoeBlock_6253472383792
// MI455X (gfx1250) — matrix operand order repaired and checked by reading, not compiled, not run on hardware
//
#include <hip/hip_runtime.h>
#include <cstdint>

// Problem constants (reference: T=512, H=2048, E=16, I=1024, top-2)
#define TNUM 512
#define HDIM 2048
#define NEXP 16
#define IDIM 1024
#define TWOI 2048
#define KT   32     // WMMA K per step (bf16)
#define KTP  40     // LDS row pitch in elements (16B-aligned chunks, conflict-avoiding)
#define MT   64     // token rows per tile pass (4 WMMA sub-tiles)

typedef __attribute__((ext_vector_type(16))) __bf16 v16bf;
typedef __attribute__((ext_vector_type(8)))  __bf16 v8bf;
typedef __attribute__((ext_vector_type(8)))  float  v8f;

#define CAT16(lo, hi) __builtin_shufflevector((lo), (hi), 0,1,2,3,4,5,6,7,8,9,10,11,12,13,14,15)

// Pack two floats into a bf16 pair: [31:16]=bf16(hi), [15:0]=bf16(lo).
// Round-half-up (+0x8000) then one v_perm_b32 byte-select.
__device__ __forceinline__ uint32_t pack2bf(float hi, float lo) {
  uint32_t a = __builtin_bit_cast(uint32_t, hi) + 0x8000u;
  uint32_t b = __builtin_bit_cast(uint32_t, lo) + 0x8000u;
  return __builtin_amdgcn_perm(a, b, 0x07060302u);  // {a[3],a[2],b[3],b[2]}
}

// ---------------------------------------------------------------------------
// Kernel 1: router. One block (256 threads) per token.
// ---------------------------------------------------------------------------
__global__ void moe_router_kernel(const float* __restrict__ x,
                                  const float* __restrict__ rw,
                                  int* __restrict__ counts,
                                  int* __restrict__ tok_idx,
                                  float* __restrict__ tok_w,
                                  int* __restrict__ tok_slot) {
  const int t = blockIdx.x;
  __shared__ float partial[NEXP][17];
  __shared__ float logits[NEXP];
  const int e    = threadIdx.x >> 4;
  const int lane = threadIdx.x & 15;
  const float* xr = x + (size_t)t * HDIM;
  const float* wr = rw + (size_t)e * HDIM;
  float s = 0.f;
  for (int h = lane; h < HDIM; h += 16) s += xr[h] * wr[h];
  partial[e][lane] = s;
  __syncthreads();
  if (threadIdx.x < NEXP) {
    float acc = 0.f;
    #pragma unroll
    for (int j = 0; j < 16; ++j) acc += partial[threadIdx.x][j];
    logits[threadIdx.x] = acc;
  }
  __syncthreads();
  if (threadIdx.x == 0) {
    float mx = -1e30f;
    #pragma unroll
    for (int i = 0; i < NEXP; ++i) mx = logits[i] > mx ? logits[i] : mx;
    float p[NEXP], psum = 0.f;
    #pragma unroll
    for (int i = 0; i < NEXP; ++i) { p[i] = __expf(logits[i] - mx); psum += p[i]; }
    float rp = __builtin_amdgcn_rcpf(psum);
    #pragma unroll
    for (int i = 0; i < NEXP; ++i) p[i] *= rp;
    int i0 = 0;
    #pragma unroll
    for (int i = 1; i < NEXP; ++i) if (p[i] > p[i0]) i0 = i;
    int i1 = (i0 == 0) ? 1 : 0;
    #pragma unroll
    for (int i = 0; i < NEXP; ++i) if (i != i0 && p[i] > p[i1]) i1 = i;
    float v0 = p[i0], v1 = p[i1];
    float inv = 1.f / (v0 + v1);
    int   ids[2] = { i0, i1 };
    float wts[2] = { v0 * inv, v1 * inv };
    #pragma unroll
    for (int k = 0; k < 2; ++k) {
      int ee = ids[k];
      int pos = atomicAdd(&counts[ee], 1);
      tok_idx [ee * TNUM + pos] = t;
      tok_w   [ee * TNUM + pos] = wts[k];
      tok_slot[ee * TNUM + pos] = k;
    }
  }
}

// ---------------------------------------------------------------------------
// Kernel 2: per-expert gathered GEMM x@w13 + SwiGLU. bf16 WMMA, f32 accum.
// grid = (IDIM/64, NEXP), block = 256 threads (8 waves).
// M-tile = 64 tokens (4 accumulators/wave), double-buffered LDS staging.
// ---------------------------------------------------------------------------
__global__ void moe_gemm13_swiglu_wmma(const float* __restrict__ x,
                                       const float* __restrict__ w13,
                                       const int* __restrict__ counts,
                                       const int* __restrict__ tok_idx,
                                       const float* __restrict__ tok_w,
                                       const int* __restrict__ tok_slot,
                                       float* __restrict__ act) {
  const int e  = blockIdx.y;
  const int c0 = blockIdx.x * 64;
  const int n_e = counts[e];
  if (n_e == 0) return;
  const int ntiles = (n_e + MT - 1) / MT;

  __shared__ __bf16 ldsA[2][MT][KTP];     // 10.0 KB
  __shared__ __bf16 ldsB[2][128][KTP];    // 20.0 KB
  __shared__ float  actG[MT][65];         // 16.25 KB
  __shared__ float  actU[MT][65];         // 16.25 KB
  __shared__ int    sTok[MT];
  __shared__ float  sW[MT];
  __shared__ int    sSlot[MT];

  const int tid  = threadIdx.x;
  const int wave = tid >> 5;
  const int lane = tid & 31;
  const float* w13e = w13 + (size_t)e * HDIM * TWOI;

  for (int tile = 0; tile < ntiles; ++tile) {
    if (tid < MT) {
      int i  = tile * MT + tid;
      int ci = (i < n_e) ? i : (n_e - 1);          // clamp: no OOB gathers
      sTok[tid]  = tok_idx [e * TNUM + ci];
      sW[tid]    = tok_w   [e * TNUM + ci];
      sSlot[tid] = tok_slot[e * TNUM + ci];
    }
    __syncthreads();

    // A: 64 rows x 32 K -> 1024 bf16-pairs, 4/thread (b64 load, b32 LDS store)
    auto stageA = [&](int buf, int k0) {
      #pragma unroll
      for (int rep = 0; rep < 4; ++rep) {
        int p  = tid + rep * 256;
        int r  = p >> 4;
        int kc = (p & 15) * 2;
        float2 f = *(const float2*)(x + (size_t)sTok[r] * HDIM + k0 + kc);
        *(uint32_t*)&ldsA[buf][r][kc] = pack2bf(f.y, f.x);
      }
    };
    // B: 32 K-rows x 128 cols, transposed to ldsB[col][k] in b32 k-pairs.
    auto stageB = [&](int buf, int k0) {
      #pragma unroll
      for (int rep = 0; rep < 2; ++rep) {
        int cb = (tid & 31) * 4;                 // col 0..124
        int kr = ((tid >> 5) + rep * 8) * 2;     // k 0,2,..,30
        int gc = (cb < 64) ? (c0 + cb) : (IDIM + c0 + (cb - 64));
        const float* s0 = w13e + (size_t)(k0 + kr) * TWOI + gc;
        float4 r0 = *(const float4*)s0;
        float4 r1 = *(const float4*)(s0 + TWOI);
        *(uint32_t*)&ldsB[buf][cb + 0][kr] = pack2bf(r1.x, r0.x);
        *(uint32_t*)&ldsB[buf][cb + 1][kr] = pack2bf(r1.y, r0.y);
        *(uint32_t*)&ldsB[buf][cb + 2][kr] = pack2bf(r1.z, r0.z);
        *(uint32_t*)&ldsB[buf][cb + 3][kr] = pack2bf(r1.w, r0.w);
      }
    };

    stageA(0, 0);
    stageB(0, 0);

    v8f acc[4] = {};
    const int nk = HDIM / KT;
    for (int ki = 0; ki < nk; ++ki) {
      const int buf = ki & 1;
      __syncthreads();
      if (ki + 1 < nk) { stageA(buf ^ 1, (ki + 1) * KT); stageB(buf ^ 1, (ki + 1) * KT); }

      const int khA  = (lane < 16) ? 0 : 8;
      const int nrow = wave * 16 + (lane & 15);
      const int khB  = (lane < 16) ? 0 : 8;
      v8bf blo = *(const v8bf*)&ldsB[buf][nrow][khB];
      v8bf bhi = *(const v8bf*)&ldsB[buf][nrow][khB + 16];
      v16bf b = CAT16(blo, bhi);
      #pragma unroll
      for (int mt = 0; mt < 4; ++mt) {
        const int m = mt * 16 + (lane & 15);
        v8bf alo = *(const v8bf*)&ldsA[buf][m][khA];
        v8bf ahi = *(const v8bf*)&ldsA[buf][m][khA + 16];
        v16bf a = CAT16(alo, ahi);
        acc[mt] = __builtin_amdgcn_wmma_f32_16x16x32_bf16(false, a, false, b,
                                                          (short)0, acc[mt], false, false);
      }
    }

    // Spill C: waves 0-3 -> gate[64][64], waves 4-7 -> up[64][64]
    {
      float (*dst)[65] = (wave < 4) ? actG : actU;
      const int csub  = (wave & 3) * 16;
      const int n     = lane & 15;
      const int mbase = (lane < 16) ? 0 : 8;
      #pragma unroll
      for (int mt = 0; mt < 4; ++mt)
        #pragma unroll
        for (int v = 0; v < 8; ++v)
          dst[mt * 16 + mbase + v][csub + n] = acc[mt][v];
    }
    __syncthreads();

    // Epilogue: act = silu(g)*u * routeW -> act[tok*2+slot][c0+c]
    #pragma unroll
    for (int rep = 0; rep < 16; ++rep) {
      int idx = tid + rep * 256;
      int r = idx >> 6;          // 0..63
      int c = idx & 63;
      int gi = tile * MT + r;
      if (gi < n_e) {
        float g = actG[r][c], u = actU[r][c];
        float sg = g * __builtin_amdgcn_rcpf(1.f + __expf(-g));
        int s = sTok[r] * 2 + sSlot[r];
        act[(size_t)s * IDIM + c0 + c] = sg * u * sW[r];
      }
    }
    __syncthreads();
  }
}

// ---------------------------------------------------------------------------
// Kernel 3: per-expert gathered GEMM act@w2 -> y[2T, H]. Same structure.
// grid = (HDIM/128, NEXP), block = 256 threads (8 waves x 16 cols).
// ---------------------------------------------------------------------------
__global__ void moe_gemm2_wmma(const float* __restrict__ act,
                               const float* __restrict__ w2,
                               const int* __restrict__ counts,
                               const int* __restrict__ tok_idx,
                               const int* __restrict__ tok_slot,
                               float* __restrict__ y) {
  const int e  = blockIdx.y;
  const int c0 = blockIdx.x * 128;
  const int n_e = counts[e];
  if (n_e == 0) return;
  const int ntiles = (n_e + MT - 1) / MT;

  __shared__ __bf16 ldsA[2][MT][KTP];
  __shared__ __bf16 ldsB[2][128][KTP];
  __shared__ int    sRow[MT];

  const int tid  = threadIdx.x;
  const int wave = tid >> 5;
  const int lane = tid & 31;
  const float* w2e = w2 + (size_t)e * IDIM * HDIM;

  for (int tile = 0; tile < ntiles; ++tile) {
    if (tid < MT) {
      int i  = tile * MT + tid;
      int ci = (i < n_e) ? i : (n_e - 1);
      sRow[tid] = tok_idx[e * TNUM + ci] * 2 + tok_slot[e * TNUM + ci];
    }
    __syncthreads();

    auto stageA = [&](int buf, int k0) {
      #pragma unroll
      for (int rep = 0; rep < 4; ++rep) {
        int p  = tid + rep * 256;
        int r  = p >> 4;
        int kc = (p & 15) * 2;
        float2 f = *(const float2*)(act + (size_t)sRow[r] * IDIM + k0 + kc);
        *(uint32_t*)&ldsA[buf][r][kc] = pack2bf(f.y, f.x);
      }
    };
    auto stageB = [&](int buf, int k0) {
      #pragma unroll
      for (int rep = 0; rep < 2; ++rep) {
        int cb = (tid & 31) * 4;
        int kr = ((tid >> 5) + rep * 8) * 2;
        const float* s0 = w2e + (size_t)(k0 + kr) * HDIM + c0 + cb;
        float4 r0 = *(const float4*)s0;
        float4 r1 = *(const float4*)(s0 + HDIM);
        *(uint32_t*)&ldsB[buf][cb + 0][kr] = pack2bf(r1.x, r0.x);
        *(uint32_t*)&ldsB[buf][cb + 1][kr] = pack2bf(r1.y, r0.y);
        *(uint32_t*)&ldsB[buf][cb + 2][kr] = pack2bf(r1.z, r0.z);
        *(uint32_t*)&ldsB[buf][cb + 3][kr] = pack2bf(r1.w, r0.w);
      }
    };

    stageA(0, 0);
    stageB(0, 0);

    v8f acc[4] = {};
    const int nk = IDIM / KT;
    for (int ki = 0; ki < nk; ++ki) {
      const int buf = ki & 1;
      __syncthreads();
      if (ki + 1 < nk) { stageA(buf ^ 1, (ki + 1) * KT); stageB(buf ^ 1, (ki + 1) * KT); }

      const int khA  = (lane < 16) ? 0 : 8;
      const int nrow = wave * 16 + (lane & 15);
      const int khB  = (lane < 16) ? 0 : 8;
      v8bf blo = *(const v8bf*)&ldsB[buf][nrow][khB];
      v8bf bhi = *(const v8bf*)&ldsB[buf][nrow][khB + 16];
      v16bf b = CAT16(blo, bhi);
      #pragma unroll
      for (int mt = 0; mt < 4; ++mt) {
        const int m = mt * 16 + (lane & 15);
        v8bf alo = *(const v8bf*)&ldsA[buf][m][khA];
        v8bf ahi = *(const v8bf*)&ldsA[buf][m][khA + 16];
        v16bf a = CAT16(alo, ahi);
        acc[mt] = __builtin_amdgcn_wmma_f32_16x16x32_bf16(false, a, false, b,
                                                          (short)0, acc[mt], false, false);
      }
    }

    // Scatter C rows to y[row][c0 + wave*16 + n]
    {
      const int n     = lane & 15;
      const int mbase = (lane < 16) ? 0 : 8;
      #pragma unroll
      for (int mt = 0; mt < 4; ++mt)
        #pragma unroll
        for (int v = 0; v < 8; ++v) {
          int r  = mt * 16 + mbase + v;
          int gi = tile * MT + r;
          if (gi < n_e)
            y[(size_t)sRow[r] * HDIM + c0 + wave * 16 + n] = acc[mt][v];
        }
    }
    __syncthreads();
  }
}

// ---------------------------------------------------------------------------
// Kernel 4: out[t] = y[2t] + y[2t+1]
// ---------------------------------------------------------------------------
__global__ void moe_combine_kernel(const float* __restrict__ y, float* __restrict__ out) {
  int idx = blockIdx.x * 256 + threadIdx.x;
  int t = idx >> 11;
  int h = idx & (HDIM - 1);
  out[idx] = y[(size_t)(2 * t) * HDIM + h] + y[(size_t)(2 * t + 1) * HDIM + h];
}

// ---------------------------------------------------------------------------
extern "C" void kernel_launch(void* const* d_in, const int* in_sizes, int n_in,
                              void* d_out, int out_size, void* d_ws, size_t ws_size,
                              hipStream_t stream) {
  const float* x   = (const float*)d_in[0];   // [512, 2048]
  const float* rw  = (const float*)d_in[1];   // [16, 2048]
  const float* w13 = (const float*)d_in[2];   // [16, 2048, 2048]
  const float* w2  = (const float*)d_in[3];   // [16, 1024, 2048]
  float* out = (float*)d_out;                 // [512, 2048]

  char* ws = (char*)d_ws;
  int*   counts   = (int*)ws;                              // 64 B
  int*   tok_idx  = (int*)(ws + 256);                      // 32 KB
  float* tok_w    = (float*)(ws + 256 + 32768);            // 32 KB
  int*   tok_slot = (int*)(ws + 256 + 2 * 32768);          // 32 KB
  float* act      = (float*)(ws + 131072);                 // [1024,1024] = 4 MB
  float* yb       = (float*)(ws + 131072 + (size_t)4 * 1024 * 1024); // [1024,2048] = 8 MB
  (void)in_sizes; (void)n_in; (void)out_size; (void)ws_size;

  hipMemsetAsync(counts, 0, NEXP * sizeof(int), stream);

  moe_router_kernel<<<TNUM, 256, 0, stream>>>(x, rw, counts, tok_idx, tok_w, tok_slot);
  moe_gemm13_swiglu_wmma<<<dim3(IDIM / 64, NEXP), 256, 0, stream>>>(
      x, w13, counts, tok_idx, tok_w, tok_slot, act);
  moe_gemm2_wmma<<<dim3(HDIM / 128, NEXP), 256, 0, stream>>>(
      act, w2, counts, tok_idx, tok_slot, yb);
  moe_combine_kernel<<<(TNUM * HDIM) / 256, 256, 0, stream>>>(yb, out);
}
